// Multi_Stage_Cross_Attention_Layer_83399674953760
// MI455X (gfx1250) — compile-verified, matrix operand order checked by reading, not run on hardware
//
#include <hip/hip_runtime.h>
#include <hip/hip_bf16.h>
#include <stdint.h>

// Problem dimensions (fixed by the reference).
#define DD   1024
#define NA   4096
#define NV   4096

// ---------------------------------------------------------------------------
// Types for CDNA5 WMMA (wave32): A = 16x32 bf16, B = 32x16 bf16, C/D = 16x16 f32
// ---------------------------------------------------------------------------
typedef __attribute__((ext_vector_type(16))) __bf16 v16bf;
typedef __attribute__((ext_vector_type(8)))  float  v8f;
typedef __attribute__((ext_vector_type(4)))  unsigned int u32x4;
typedef __attribute__((ext_vector_type(8)))  unsigned int u32x8;

__device__ __forceinline__ unsigned short f32_to_bf16_bits(float f) {
  unsigned u = __float_as_uint(f);
  return (unsigned short)((u + 0x7FFFu + ((u >> 16) & 1u)) >> 16);  // RNE
}

struct Frag {
  union { v16bf v; uint4 q[2]; };
};

// Load a 16x32 bf16 fragment (A: rows = M, B: rows = N), K-contiguous source.
// ISA layout: lanes 0-15 hold row r with K-chunk [half*8 .. half*8+7] in v[0..7]
// and K-chunk [16+half*8 ..] in v[8..15], half = lane>>4.
__device__ __forceinline__ Frag load_frag(const unsigned short* __restrict__ base,
                                          int ld, int row, int k, int lane) {
  Frag f;
  const unsigned short* p =
      base + (size_t)(row + (lane & 15)) * ld + k + ((lane >> 4) << 3);
  f.q[0] = *reinterpret_cast<const uint4*>(p);
  f.q[1] = *reinterpret_cast<const uint4*>(p + 16);
  return f;
}

// ---------------------------------------------------------------------------
// Tensor Data Mover: 2D tile (tile_d1 rows x tile_d0 elems) of 2-byte data,
// global -> LDS. Descriptor built per CDNA5 ISA.
// Inline asm (portable across toolchains; builtin arity differs).
// ---------------------------------------------------------------------------
__device__ __forceinline__ void tdm_load_2d_bf16_to_lds(
    const void* gaddr, unsigned lds_byte_off,
    unsigned tensor_d0, unsigned tensor_d1,
    unsigned tile_d0, unsigned tile_d1, unsigned stride_elems) {
  const unsigned long long ga = (unsigned long long)(uintptr_t)gaddr;
  u32x4 g0;
  g0[0] = 1u;                                   // count=1, user descriptor
  g0[1] = lds_byte_off;                         // lds_addr (bytes)
  g0[2] = (unsigned)ga;                         // global_addr[31:0]
  g0[3] = (unsigned)(ga >> 32) | (2u << 30);    // global_addr[56:32] | type=2
  u32x8 g1;
  g1[0] = (1u << 16);                           // wg_mask=0, data_size=1 (2B)
  g1[1] = (tensor_d0 & 0xFFFFu) << 16;          // atomic_bar=0; dim0[15:0]
  g1[2] = (tensor_d0 >> 16) | ((tensor_d1 & 0xFFFFu) << 16);  // dim0[31:16], dim1[15:0]
  g1[3] = (tensor_d1 >> 16) | (tile_d0 << 16);  // dim1[31:16], tile_dim0
  g1[4] = tile_d1;                              // tile_dim1 (tile_dim2 = 0)
  g1[5] = stride_elems;                         // tensor_dim0_stride[31:0]
  g1[6] = 0;                                    // stride[47:32], dim1_stride[15:0]
  g1[7] = 0;                                    // dim1_stride[47:16]
  asm volatile("tensor_load_to_lds %0, %1" :: "s"(g0), "s"(g1) : "memory");
}

// ---------------------------------------------------------------------------
// Generic GEMM: C[M,N] = A[M,K] @ B[N,K]^T  (bf16 operands, K-contiguous)
// One wave computes a 64x64 tile: 4x4 grid of 16x16 WMMA accumulators.
// OUT_MODE: 0 = store bf16, 1 = store f32, 2 = store f32 + residual (fp32)
// ---------------------------------------------------------------------------
template <int OUT_MODE>
__global__ __launch_bounds__(128) void gemm_bf16_tn(
    const unsigned short* __restrict__ A, const unsigned short* __restrict__ B,
    void* __restrict__ Cout, const float* __restrict__ residual,
    int M, int N, int K) {
  const int lane = threadIdx.x & 31;
  const int wave = threadIdx.x >> 5;
  const int tilesN = N >> 6;
  const int tilesM = M >> 6;
  const int t = blockIdx.x * 4 + wave;
  if (t >= tilesM * tilesN) return;
  const int m0 = (t / tilesN) << 6;
  const int n0 = (t % tilesN) << 6;

  v8f acc[4][4] = {};

  for (int kb = 0; kb < K; kb += 32) {
    Frag a[4], b[4];
#pragma unroll
    for (int i = 0; i < 4; ++i) a[i] = load_frag(A, K, m0 + 16 * i, kb, lane);
#pragma unroll
    for (int j = 0; j < 4; ++j) b[j] = load_frag(B, K, n0 + 16 * j, kb, lane);
#pragma unroll
    for (int i = 0; i < 4; ++i)
#pragma unroll
      for (int j = 0; j < 4; ++j)
        acc[i][j] = __builtin_amdgcn_wmma_f32_16x16x32_bf16(
            false, a[i].v, false, b[j].v, (short)0, acc[i][j], false, false);
  }

  // C/D layout: VGPR r, lanes 0-15 -> M = r, lanes 16-31 -> M = 8 + r; N = lane&15
  const int nIdx = lane & 15;
  const int half = lane >> 4;
#pragma unroll
  for (int i = 0; i < 4; ++i)
#pragma unroll
    for (int j = 0; j < 4; ++j) {
      const int col = n0 + 16 * j + nIdx;
#pragma unroll
      for (int r = 0; r < 8; ++r) {
        const int row = m0 + 16 * i + 8 * half + r;
        const size_t idx = (size_t)row * N + col;
        const float val = acc[i][j][r];
        if (OUT_MODE == 0) {
          ((unsigned short*)Cout)[idx] = f32_to_bf16_bits(val);
        } else if (OUT_MODE == 1) {
          ((float*)Cout)[idx] = val;
        } else {
          ((float*)Cout)[idx] = val + residual[idx];
        }
      }
    }
}

// ---------------------------------------------------------------------------
// Projection GEMM with TDM-staged A operand.
// Block = 4 waves covering 64 (M) x 256 (N); all waves share the same 64 M-rows.
// Per k-step: wave 0 DMAs A[64 x 32] bf16 (4 KB) into LDS via the Tensor Data
// Mover, waits TENSORcnt, block barriers; all waves read A fragments from LDS
// (ds_load_b128) and B fragments from global. Output stored bf16.
// ---------------------------------------------------------------------------
__global__ __launch_bounds__(128) void gemm_proj_tdm(
    const unsigned short* __restrict__ A, const unsigned short* __restrict__ B,
    unsigned short* __restrict__ Cout, int M, int N, int K) {
  __shared__ unsigned short Atile[64 * 32];  // 4 KB staging buffer

  const int lane = threadIdx.x & 31;
  const int wave = threadIdx.x >> 5;
  const int nBlocks = N >> 8;                 // 256-wide N blocks
  const int m0 = (blockIdx.x / nBlocks) << 6;
  const int n0 = ((blockIdx.x % nBlocks) << 8) + (wave << 6);

  const unsigned lds_off = (unsigned)(uintptr_t)(&Atile[0]);  // LDS aperture: low 32 bits = offset

  v8f acc[4][4] = {};

  for (int kb = 0; kb < K; kb += 32) {
    if (wave == 0) {
      tdm_load_2d_bf16_to_lds(A + (size_t)m0 * K + kb, lds_off,
                              /*tensor_d0=*/(unsigned)K, /*tensor_d1=*/64u,
                              /*tile_d0=*/32u, /*tile_d1=*/64u,
                              /*stride_elems=*/(unsigned)K);
      __builtin_amdgcn_s_wait_tensorcnt(0);
    }
    __syncthreads();  // publish TDM-written tile to all waves

    Frag a[4], b[4];
#pragma unroll
    for (int i = 0; i < 4; ++i) {
      const unsigned short* p =
          &Atile[(16 * i + (lane & 15)) * 32 + ((lane >> 4) << 3)];
      a[i].q[0] = *reinterpret_cast<const uint4*>(p);        // ds_load_b128
      a[i].q[1] = *reinterpret_cast<const uint4*>(p + 16);
    }
#pragma unroll
    for (int j = 0; j < 4; ++j) b[j] = load_frag(B, K, n0 + 16 * j, kb, lane);
#pragma unroll
    for (int i = 0; i < 4; ++i)
#pragma unroll
      for (int j = 0; j < 4; ++j)
        acc[i][j] = __builtin_amdgcn_wmma_f32_16x16x32_bf16(
            false, a[i].v, false, b[j].v, (short)0, acc[i][j], false, false);

    __syncthreads();  // all waves done reading before next TDM overwrite
  }

  const int nIdx = lane & 15;
  const int half = lane >> 4;
#pragma unroll
  for (int i = 0; i < 4; ++i)
#pragma unroll
    for (int j = 0; j < 4; ++j) {
      const int col = n0 + 16 * j + nIdx;
#pragma unroll
      for (int r = 0; r < 8; ++r) {
        const int row = m0 + 16 * i + 8 * half + r;
        Cout[(size_t)row * N + col] = f32_to_bf16_bits(acc[i][j][r]);
      }
    }
}

// ---------------------------------------------------------------------------
// Elementwise fp32 -> bf16 convert
// ---------------------------------------------------------------------------
__global__ __launch_bounds__(256) void cvt_f32_bf16(
    const float* __restrict__ src, unsigned short* __restrict__ dst, size_t n) {
  size_t i = (size_t)blockIdx.x * blockDim.x + threadIdx.x;
  size_t stride = (size_t)gridDim.x * blockDim.x;
  for (; i < n; i += stride) dst[i] = f32_to_bf16_bits(src[i]);
}

// ---------------------------------------------------------------------------
// Tiled transpose: fp32 [R,C] -> bf16 [C,R]   (R, C multiples of 32)
// ---------------------------------------------------------------------------
__global__ __launch_bounds__(256) void transpose_f32_to_bf16(
    const float* __restrict__ src, unsigned short* __restrict__ dst, int R, int C) {
  __shared__ float tile[32][33];
  const int tilesC = C >> 5;
  const int c0 = (blockIdx.x % tilesC) << 5;
  const int r0 = (blockIdx.x / tilesC) << 5;
  const int tx = threadIdx.x & 31;
  const int ty = threadIdx.x >> 5;  // 0..7
#pragma unroll
  for (int i = 0; i < 32; i += 8)
    tile[ty + i][tx] = src[(size_t)(r0 + ty + i) * C + c0 + tx];
  __syncthreads();
#pragma unroll
  for (int i = 0; i < 32; i += 8)
    dst[(size_t)(c0 + ty + i) * R + r0 + tx] = f32_to_bf16_bits(tile[tx][ty + i]);
}

// Tiled transpose: bf16 [R,C] -> bf16 [C,R]
__global__ __launch_bounds__(256) void transpose_bf16(
    const unsigned short* __restrict__ src, unsigned short* __restrict__ dst,
    int R, int C) {
  __shared__ unsigned short tile[32][33];
  const int tilesC = C >> 5;
  const int c0 = (blockIdx.x % tilesC) << 5;
  const int r0 = (blockIdx.x / tilesC) << 5;
  const int tx = threadIdx.x & 31;
  const int ty = threadIdx.x >> 5;
#pragma unroll
  for (int i = 0; i < 32; i += 8)
    tile[ty + i][tx] = src[(size_t)(r0 + ty + i) * C + c0 + tx];
  __syncthreads();
#pragma unroll
  for (int i = 0; i < 32; i += 8)
    dst[(size_t)(c0 + ty + i) * R + r0 + tx] = tile[tx][ty + i];
}

// ---------------------------------------------------------------------------
// Row softmax, in place, fp32. One 256-thread block (8 waves) per row.
// ---------------------------------------------------------------------------
__global__ __launch_bounds__(256) void softmax_rows(float* __restrict__ S, int N) {
  float* p = S + (size_t)blockIdx.x * N;
  __shared__ float red[8];
  const int lane = threadIdx.x & 31;
  const int wid = threadIdx.x >> 5;

  float mx = -INFINITY;
  for (int i = threadIdx.x; i < N; i += 256) mx = fmaxf(mx, p[i]);
#pragma unroll
  for (int o = 16; o > 0; o >>= 1) mx = fmaxf(mx, __shfl_xor(mx, o, 32));
  if (lane == 0) red[wid] = mx;
  __syncthreads();
  float rmax = red[0];
#pragma unroll
  for (int w = 1; w < 8; ++w) rmax = fmaxf(rmax, red[w]);
  __syncthreads();

  float sum = 0.0f;
  for (int i = threadIdx.x; i < N; i += 256) {
    float e = __expf(p[i] - rmax);
    p[i] = e;
    sum += e;
  }
#pragma unroll
  for (int o = 16; o > 0; o >>= 1) sum += __shfl_xor(sum, o, 32);
  if (lane == 0) red[wid] = sum;
  __syncthreads();
  float rsum = 0.0f;
#pragma unroll
  for (int w = 0; w < 8; ++w) rsum += red[w];
  const float inv = 1.0f / rsum;
  for (int i = threadIdx.x; i < N; i += 256) p[i] *= inv;
}

// ---------------------------------------------------------------------------
// Orchestration
// ---------------------------------------------------------------------------
static void launch_gemm(int mode, const unsigned short* A, const unsigned short* B,
                        void* C, const float* res, int M, int N, int K,
                        hipStream_t s) {
  const int tiles = (M >> 6) * (N >> 6);
  const int blocks = tiles >> 2;  // 4 waves per block, all dims multiples of 64
  switch (mode) {
    case 0: gemm_bf16_tn<0><<<blocks, 128, 0, s>>>(A, B, C, res, M, N, K); break;
    case 1: gemm_bf16_tn<1><<<blocks, 128, 0, s>>>(A, B, C, res, M, N, K); break;
    default: gemm_bf16_tn<2><<<blocks, 128, 0, s>>>(A, B, C, res, M, N, K); break;
  }
}

static void launch_proj_tdm(const unsigned short* A, const unsigned short* B,
                            unsigned short* C, int M, int N, int K, hipStream_t s) {
  const int blocks = (M >> 6) * (N >> 8);  // 64 x 256 per block
  gemm_proj_tdm<<<blocks, 128, 0, s>>>(A, B, C, M, N, K);
}

extern "C" void kernel_launch(void* const* d_in, const int* in_sizes, int n_in,
                              void* d_out, int out_size, void* d_ws, size_t ws_size,
                              hipStream_t stream) {
  (void)in_sizes; (void)n_in; (void)out_size; (void)ws_size;

  // Input order from setup_inputs(): inputs_a, inputs_v, W_Ba, W_Aa, W_Bv, W_Av
  const float* inputs_a = (const float*)d_in[0];
  const float* inputs_v = (const float*)d_in[1];
  const float* W_Ba     = (const float*)d_in[2];
  const float* W_Aa     = (const float*)d_in[3];
  const float* W_Bv     = (const float*)d_in[4];
  const float* W_Av     = (const float*)d_in[5];
  float* out = (float*)d_out;  // [feature_map_a | feature_map_v], each NA*DD f32

  // Workspace carve-up (~200 MB)
  char* ws = (char*)d_ws;
  auto carve = [&](size_t bytes) -> char* {
    char* p = ws;
    ws += (bytes + 255) & ~(size_t)255;
    return p;
  };
  unsigned short* Xa_bf   = (unsigned short*)carve((size_t)NA * DD * 2);
  unsigned short* Xv_bf   = (unsigned short*)carve((size_t)NV * DD * 2);
  unsigned short* WBa_bf  = (unsigned short*)carve((size_t)DD * DD * 2);
  unsigned short* WAa_bf  = (unsigned short*)carve((size_t)DD * DD * 2);
  unsigned short* WBv_bf  = (unsigned short*)carve((size_t)DD * DD * 2);
  unsigned short* WAv_bf  = (unsigned short*)carve((size_t)DD * DD * 2);
  unsigned short* aa_bf   = (unsigned short*)carve((size_t)NA * DD * 2);
  unsigned short* ba_bf   = (unsigned short*)carve((size_t)NA * DD * 2);
  unsigned short* av_bf   = (unsigned short*)carve((size_t)NV * DD * 2);
  unsigned short* bv_bf   = (unsigned short*)carve((size_t)NV * DD * 2);
  unsigned short* aaT_bf  = (unsigned short*)carve((size_t)DD * NA * 2);
  unsigned short* bvT_bf  = (unsigned short*)carve((size_t)DD * NV * 2);
  unsigned short* alpha_bf  = (unsigned short*)carve((size_t)NA * NV * 2);
  unsigned short* alphaT_bf = (unsigned short*)carve((size_t)NV * NA * 2);
  float* scores = (float*)carve((size_t)NA * NV * 4);  // alpha in place after softmax

  // 1) Convert inputs + weights to bf16
  const size_t nX = (size_t)NA * DD, nW = (size_t)DD * DD;
  cvt_f32_bf16<<<4096, 256, 0, stream>>>(inputs_a, Xa_bf, nX);
  cvt_f32_bf16<<<4096, 256, 0, stream>>>(inputs_v, Xv_bf, nX);
  cvt_f32_bf16<<<1024, 256, 0, stream>>>(W_Ba, WBa_bf, nW);
  cvt_f32_bf16<<<1024, 256, 0, stream>>>(W_Aa, WAa_bf, nW);
  cvt_f32_bf16<<<1024, 256, 0, stream>>>(W_Bv, WBv_bf, nW);
  cvt_f32_bf16<<<1024, 256, 0, stream>>>(W_Av, WAv_bf, nW);

  // 2) Projections: y = x @ W^T  (TN GEMM, A staged via Tensor Data Mover)
  launch_proj_tdm(Xa_bf, WAa_bf, aa_bf, NA, DD, DD, stream);
  launch_proj_tdm(Xa_bf, WBa_bf, ba_bf, NA, DD, DD, stream);
  launch_proj_tdm(Xv_bf, WAv_bf, av_bf, NV, DD, DD, stream);
  launch_proj_tdm(Xv_bf, WBv_bf, bv_bf, NV, DD, DD, stream);

  // 3) scores = b_a @ a_v^T  (TN: both K=D contiguous), f32 output
  launch_gemm(1, ba_bf, av_bf, scores, nullptr, NA, NV, DD, stream);

  // 4) softmax over N_v, in place
  softmax_rows<<<NA, 256, 0, stream>>>(scores, NV);

  // 5) Prepare operands for the attention-apply GEMMs
  cvt_f32_bf16<<<8192, 256, 0, stream>>>(scores, alpha_bf, (size_t)NA * NV);
  transpose_f32_to_bf16<<<(NA / 32) * (NV / 32), 256, 0, stream>>>(scores, alphaT_bf, NA, NV);
  transpose_bf16<<<(NA / 32) * (DD / 32), 256, 0, stream>>>(aa_bf, aaT_bf, NA, DD);
  transpose_bf16<<<(NV / 32) * (DD / 32), 256, 0, stream>>>(bv_bf, bvT_bf, NV, DD);

  // 6) feature_map_a = alpha @ b_v + inputs_a
  //    C[a,d] = sum_v alpha[a,v] * bvT[d,v]  -> TN with K = N_v
  launch_gemm(2, alpha_bf, bvT_bf, out, inputs_a, NA, DD, NV, stream);

  // 7) feature_map_v = alpha^T @ a_a + inputs_v
  //    C[v,d] = sum_a alphaT[v,a] * aaT[d,a] -> TN with K = N_a
  launch_gemm(2, alphaT_bf, aaT_bf, out + (size_t)NA * DD, inputs_v, NV, DD, NA, stream);
}
